// TransformerGNN_84207128805751
// MI455X (gfx1250) — matrix operand order repaired and checked by reading, not compiled, not run on hardware
//
#include <hip/hip_runtime.h>
#include <math.h>

typedef __attribute__((ext_vector_type(16))) __bf16 v16bf;
typedef __attribute__((ext_vector_type(8)))  float  v8f;

#define N_NODES 20000
#define N_EDGES 320000
#define HID 256

static __device__ __forceinline__ float geluf(float x) {
    return 0.5f * x * (1.0f + erff(x * 0.70710678118654752f));
}

static __device__ __forceinline__ void atomicMaxF(float* addr, float val) {
    if (val >= 0.0f) atomicMax((int*)addr, __float_as_int(val));
    else             atomicMin((unsigned int*)addr, __float_as_uint(val));
}

// ---------------- elementwise kernels ----------------
__global__ void k_fill(float* p, float v, size_t n) {
    for (size_t i = (size_t)blockIdx.x * blockDim.x + threadIdx.x; i < n;
         i += (size_t)gridDim.x * blockDim.x) p[i] = v;
}

__global__ void k_add(float* o, const float* a, const float* b, size_t n) {
    for (size_t i = (size_t)blockIdx.x * blockDim.x + threadIdx.x; i < n;
         i += (size_t)gridDim.x * blockDim.x) o[i] = a[i] + b[i];
}

// pad/convert fp32 [sr,sc] -> bf16 [dr,dc] (zero-padded)
__global__ void k_pad_bf16(__bf16* d, const float* s, int dr, int dc, int sr, int sc) {
    size_t n = (size_t)dr * dc;
    for (size_t i = (size_t)blockIdx.x * blockDim.x + threadIdx.x; i < n;
         i += (size_t)gridDim.x * blockDim.x) {
        int r = (int)(i / dc), c = (int)(i % dc);
        d[i] = (r < sr && c < sc) ? (__bf16)s[(size_t)r * sc + c] : (__bf16)0.0f;
    }
}

// ---------------- WMMA GEMM: C[M,Nn] = A[M,K](bf16) @ B[K,Nn](bf16) ----------------
// block = 256 threads (8 waves), tile 128x64, BK=32. wave w -> rows [w*16, w*16+16)
// LDS holds tiles pre-swizzled into fragment-linear order:
//   ldsA[(w*32+lane)*16 + p] : wave w's A fragment, lane-major, 32B/lane contiguous
//   ldsB[(nt*32+lane)*16 + p]: N-subtile nt's B fragment, lane-major, 32B/lane contiguous
// so fragment reads are single v16bf (2x ds_load_b128) loads; all 5 fragments are
// loaded before the 4 WMMAs so the LDS loads batch and overlap the matrix pipe.
__global__ __launch_bounds__(256) void k_gemm(const __bf16* __restrict__ A, int K,
                                              const __bf16* __restrict__ B, int Nn, int M,
                                              const float* __restrict__ bias,
                                              const float* __restrict__ resid,
                                              float* __restrict__ Cf, __bf16* __restrict__ Cb,
                                              int act) {
    __shared__ __align__(32) __bf16 lA[8 * 32 * 16];   // 8KB
    __shared__ __align__(32) __bf16 lB[4 * 32 * 16];   // 4KB
    const int tid = threadIdx.x, lane = tid & 31, wave = tid >> 5;
    const int m0 = blockIdx.x * 128, n0 = blockIdx.y * 64;
    v8f acc[4] = {};

    for (int kk = 0; kk < K; kk += 32) {
        // ---- A tile 128x32: 512 chunks of 8 bf16; 2 chunks per thread ----
        // swizzle: (r, cg) -> lane = (r&15) + (cg&1)*16, wave = r>>4, pHalf = cg>>1
        #pragma unroll
        for (int j = 0; j < 2; ++j) {
            int chunk = tid * 2 + j;
            int r = chunk >> 2, cg = chunk & 3;
            int gr = m0 + r;
            uint4 av = make_uint4(0, 0, 0, 0);
            if (gr < M) {
                const __bf16* gp = A + (size_t)gr * K + kk + cg * 8;
                av = *(const uint4*)gp;
                if (kk + 32 < K) __builtin_prefetch(gp + 32, 0, 1);
            }
            int dlane = (r & 15) + ((cg & 1) << 4);
            int dw = r >> 4, pHalf = cg >> 1;
            *(uint4*)(lA + ((dw * 32 + dlane) * 16 + pHalf * 8)) = av;
        }
        // ---- B tile 32x64: 256 chunks of 8 bf16; 1 chunk per thread ----
        {
            int r = tid >> 3, c0 = (tid & 7) * 8;
            union { uint4 u; __bf16 h[8]; } bv;
            bv.u = *(const uint4*)(B + (size_t)(kk + r) * Nn + n0 + c0);
            int laneHi = ((r >> 3) & 1) << 4;   // K bit3 -> +16 lanes
            int p = (r & 7) + ((r >> 4) << 3);
            #pragma unroll
            for (int jj = 0; jj < 8; ++jj) {
                int c = c0 + jj;
                int nt = c >> 4;
                int dlane = (c & 15) + laneHi;
                lB[(nt * 32 + dlane) * 16 + p] = bv.h[jj];
            }
        }
        __syncthreads();

        // batch all fragment loads, then issue 4 independent WMMAs back-to-back
        v16bf af  = *(const v16bf*)(lA + (wave * 32 + lane) * 16);
        v16bf bf0 = *(const v16bf*)(lB + (0 * 32 + lane) * 16);
        v16bf bf1 = *(const v16bf*)(lB + (1 * 32 + lane) * 16);
        v16bf bf2 = *(const v16bf*)(lB + (2 * 32 + lane) * 16);
        v16bf bf3 = *(const v16bf*)(lB + (3 * 32 + lane) * 16);
        acc[0] = __builtin_amdgcn_wmma_f32_16x16x32_bf16(false, af, false, bf0, (short)0, acc[0], false, false);
        acc[1] = __builtin_amdgcn_wmma_f32_16x16x32_bf16(false, af, false, bf1, (short)0, acc[1], false, false);
        acc[2] = __builtin_amdgcn_wmma_f32_16x16x32_bf16(false, af, false, bf2, (short)0, acc[2], false, false);
        acc[3] = __builtin_amdgcn_wmma_f32_16x16x32_bf16(false, af, false, bf3, (short)0, acc[3], false, false);
        __syncthreads();
    }

    // C/D layout: VGPR i -> row i (lanes 0-15) / i+8 (lanes 16-31), col = lane&15
    const int colLo = lane & 15, halfRow = (lane >> 4) * 8;
    #pragma unroll
    for (int nt = 0; nt < 4; ++nt) {
        int col = n0 + nt * 16 + colLo;
        #pragma unroll
        for (int i = 0; i < 8; ++i) {
            int row = m0 + wave * 16 + halfRow + i;
            if (row < M) {
                float v = acc[nt][i];
                if (bias)  v += bias[col];
                if (resid) v += resid[(size_t)row * Nn + col];
                if (act)   v = geluf(v);
                if (Cf) Cf[(size_t)row * Nn + col] = v;
                if (Cb) Cb[(size_t)row * Nn + col] = (__bf16)v;
            }
        }
    }
}

// ---------------- LayerNorm (+GELU), one wave per row ----------------
__global__ void k_ln(const float* __restrict__ x, const float* __restrict__ g,
                     const float* __restrict__ b, float* __restrict__ of,
                     __bf16* __restrict__ obf, int rows, int cols, int doGelu) {
    int wave = blockIdx.x * (blockDim.x >> 5) + (threadIdx.x >> 5);
    int lane = threadIdx.x & 31;
    if (wave >= rows) return;
    const float* xr = x + (size_t)wave * cols;
    float s = 0.0f;
    for (int c = lane; c < cols; c += 32) s += xr[c];
    for (int o = 16; o > 0; o >>= 1) s += __shfl_xor(s, o, 32);
    float mean = s / cols;
    float v = 0.0f;
    for (int c = lane; c < cols; c += 32) { float d = xr[c] - mean; v += d * d; }
    for (int o = 16; o > 0; o >>= 1) v += __shfl_xor(v, o, 32);
    float inv = rsqrtf(v / cols + 1e-5f);
    for (int c = lane; c < cols; c += 32) {
        float y = (xr[c] - mean) * inv * g[c] + b[c];
        if (doGelu) y = geluf(y);
        if (of)  of[(size_t)wave * cols + c] = y;
        if (obf) obf[(size_t)wave * cols + c] = (__bf16)y;
    }
}

// ---------------- edge projection: e[E,32] = edge_attr[E,1] * ep_W[1,32] + ep_b ----------------
__global__ void k_edge_proj(const float* __restrict__ ea, const float* __restrict__ W,
                            const float* __restrict__ b, float* __restrict__ e) {
    size_t n = (size_t)N_EDGES * 32;
    for (size_t i = (size_t)blockIdx.x * blockDim.x + threadIdx.x; i < n;
         i += (size_t)gridDim.x * blockDim.x) {
        int ed = (int)(i >> 5), c = (int)(i & 31);
        e[i] = ea[ed] * W[c] + b[c];
    }
}

// ---------------- attention pass 1: logits + segment max ----------------
__global__ void k_edge_logits(const float* __restrict__ q, const float* __restrict__ kx,
                              const float* __restrict__ eproj, const float* __restrict__ We,
                              const int* __restrict__ src, const int* __restrict__ dst,
                              int H, int C, float* __restrict__ logits, float* __restrict__ mMax) {
    int total = N_EDGES * H;
    float scale = 1.0f / sqrtf((float)C);
    for (int i = blockIdx.x * blockDim.x + threadIdx.x; i < total;
         i += gridDim.x * blockDim.x) {
        int e = i / H, h = i - e * H;
        int s = src[e], d = dst[e];
        const float* er = eproj + (size_t)e * 32;
        float acc = 0.0f;
        for (int c = 0; c < C; ++c) {
            int col = h * C + c;
            float ee = 0.0f;
            #pragma unroll
            for (int j = 0; j < 32; ++j) ee += er[j] * We[j * HID + col];
            acc += q[(size_t)d * HID + col] * (kx[(size_t)s * HID + col] + ee);
        }
        float lg = acc * scale;
        logits[i] = lg;
        atomicMaxF(&mMax[d * H + h], lg);
    }
}

// ---------------- attention pass 2: exp + segment sum ----------------
__global__ void k_edge_expsum(float* __restrict__ logits, const float* __restrict__ mMax,
                              const int* __restrict__ dst, float* __restrict__ den, int H) {
    int total = N_EDGES * H;
    for (int i = blockIdx.x * blockDim.x + threadIdx.x; i < total;
         i += gridDim.x * blockDim.x) {
        int e = i / H, h = i - e * H;
        int d = dst[e];
        float ex = expf(logits[i] - mMax[d * H + h]);
        logits[i] = ex;
        atomicAdd(&den[d * H + h], ex);
    }
}

// ---------------- attention pass 3: weighted scatter-add ----------------
__global__ void k_edge_agg(const float* __restrict__ v, const float* __restrict__ eproj,
                           const float* __restrict__ We, const int* __restrict__ src,
                           const int* __restrict__ dst, const float* __restrict__ ex,
                           const float* __restrict__ den, float* __restrict__ agg,
                           int H, int C) {
    size_t total = (size_t)N_EDGES * HID;
    for (size_t i = (size_t)blockIdx.x * blockDim.x + threadIdx.x; i < total;
         i += (size_t)gridDim.x * blockDim.x) {
        int e = (int)(i >> 8), c = (int)(i & 255);
        int h = c / C;
        int s = src[e], d = dst[e];
        float w = ex[(size_t)e * H + h] / (den[d * H + h] + 1e-16f);
        const float* er = eproj + (size_t)e * 32;
        float ee = 0.0f;
        #pragma unroll
        for (int j = 0; j < 32; ++j) ee += er[j] * We[j * HID + c];
        atomicAdd(&agg[(size_t)d * HID + c], (v[(size_t)s * HID + c] + ee) * w);
    }
}

// ---------------- gated skip (beta): one block per node ----------------
__global__ void k_beta(const float* __restrict__ out, const float* __restrict__ xr,
                       const float* __restrict__ Wb, float* __restrict__ h) {
    __shared__ float red[256];
    int n = blockIdx.x, c = threadIdx.x;
    float o = out[(size_t)n * HID + c], x = xr[(size_t)n * HID + c];
    red[c] = o * Wb[c] + x * Wb[HID + c] + (o - x) * Wb[2 * HID + c];
    __syncthreads();
    for (int s = 128; s > 0; s >>= 1) {
        if (c < s) red[c] += red[c + s];
        __syncthreads();
    }
    float b = 1.0f / (1.0f + expf(-red[0]));
    h[(size_t)n * HID + c] = b * x + (1.0f - b) * o;
}

// ---------------- final head: out[n] = dot(om[n,0:128], W) + b, one wave per node ----------------
__global__ void k_out(const float* __restrict__ om, const float* __restrict__ W,
                      const float* __restrict__ b, float* __restrict__ out) {
    int wave = (blockIdx.x * blockDim.x + threadIdx.x) >> 5;
    int lane = threadIdx.x & 31;
    if (wave >= N_NODES) return;
    float s = 0.0f;
    for (int c = lane; c < 128; c += 32) s += om[(size_t)wave * 128 + c] * W[c];
    for (int o = 16; o > 0; o >>= 1) s += __shfl_xor(s, o, 32);
    if (lane == 0) out[wave] = s + b[0];
}

extern "C" void kernel_launch(void* const* d_in, const int* in_sizes, int n_in,
                              void* d_out, int out_size, void* d_ws, size_t ws_size,
                              hipStream_t stream) {
    (void)in_sizes; (void)n_in; (void)out_size; (void)ws_size;
    const float* x  = (const float*)d_in[0];
    const int*   ei = (const int*)d_in[1];
    const float* ea = (const float*)d_in[2];
    auto F = [&](int i) { return (const float*)d_in[i]; };
    const int* src = ei;
    const int* dst = ei + N_EDGES;

    // bump allocator over d_ws
    char* base = (char*)d_ws; size_t off = 0;
    auto alloc = [&](size_t bytes) -> void* {
        void* p = base + off; off = (off + bytes + 255) & ~(size_t)255; return p;
    };

    __bf16* xbf  = (__bf16*)alloc((size_t)N_NODES * 32 * 2);
    __bf16* hbf  = (__bf16*)alloc((size_t)N_NODES * HID * 2);
    __bf16* ffbf = (__bf16*)alloc((size_t)N_NODES * 1024 * 2);
    __bf16* wscr = (__bf16*)alloc((size_t)1024 * 1024 * 2);     // weight conversion scratch
    float* qb  = (float*)alloc((size_t)N_NODES * HID * 4);
    float* kb  = (float*)alloc((size_t)N_NODES * HID * 4);
    float* vb  = (float*)alloc((size_t)N_NODES * HID * 4);
    float* xr  = (float*)alloc((size_t)N_NODES * HID * 4);
    float* res = (float*)alloc((size_t)N_NODES * HID * 4);
    float* agg = (float*)alloc((size_t)N_NODES * HID * 4);
    float* t0  = (float*)alloc((size_t)N_NODES * HID * 4);
    float* t1  = (float*)alloc((size_t)N_NODES * HID * 4);
    float* hf  = (float*)alloc((size_t)N_NODES * HID * 4);
    float* eproj = (float*)alloc((size_t)N_EDGES * 32 * 4);
    float* exb   = (float*)alloc((size_t)N_EDGES * 8 * 4);
    float* mMax  = (float*)alloc((size_t)N_NODES * 8 * 4);
    float* den   = (float*)alloc((size_t)N_NODES * 8 * 4);

    const dim3 gN((N_NODES + 127) / 128, HID / 64);       // GEMM -> 256-wide output
    const int lnB = (N_NODES + 7) / 8;

    // input projection: pad x -> bf16 [N,32], pad in_W -> [32,256], GEMM, LN+GELU
    k_pad_bf16<<<4096, 256, 0, stream>>>(xbf, x, N_NODES, 32, N_NODES, 31);
    k_pad_bf16<<<256, 256, 0, stream>>>(wscr, F(3), 32, 256, 31, 256);
    k_gemm<<<gN, 256, 0, stream>>>(xbf, 32, wscr, 256, N_NODES, F(4), nullptr, t0, nullptr, 0);
    k_ln<<<lnB, 256, 0, stream>>>(t0, F(5), F(6), hf, hbf, N_NODES, 256, 1);

    // edge projection
    k_edge_proj<<<4096, 256, 0, stream>>>(ea, F(7), F(8), eproj);

    for (int L = 0; L < 3; ++L) {
        const int bi = 9 + L * 16;
        const int H = (L == 2) ? 1 : 8;
        const int C = HID / H;
        const float* We = F(bi + 6);

        if (L < 2)
            hipMemcpyAsync(res, hf, (size_t)N_NODES * HID * 4,
                           hipMemcpyDeviceToDevice, stream);

        // Q, K, V, skip projections (bf16 WMMA GEMMs)
        k_pad_bf16<<<512, 256, 0, stream>>>(wscr, F(bi + 0), 256, 256, 256, 256);
        k_gemm<<<gN, 256, 0, stream>>>(hbf, 256, wscr, 256, N_NODES, F(bi + 1), nullptr, qb, nullptr, 0);
        k_pad_bf16<<<512, 256, 0, stream>>>(wscr, F(bi + 2), 256, 256, 256, 256);
        k_gemm<<<gN, 256, 0, stream>>>(hbf, 256, wscr, 256, N_NODES, F(bi + 3), nullptr, kb, nullptr, 0);
        k_pad_bf16<<<512, 256, 0, stream>>>(wscr, F(bi + 4), 256, 256, 256, 256);
        k_gemm<<<gN, 256, 0, stream>>>(hbf, 256, wscr, 256, N_NODES, F(bi + 5), nullptr, vb, nullptr, 0);
        k_pad_bf16<<<512, 256, 0, stream>>>(wscr, F(bi + 7), 256, 256, 256, 256);
        k_gemm<<<gN, 256, 0, stream>>>(hbf, 256, wscr, 256, N_NODES, F(bi + 8), nullptr, xr, nullptr, 0);

        // segment softmax + scatter aggregation
        k_fill<<<64, 256, 0, stream>>>(mMax, -1e30f, (size_t)N_NODES * H);
        k_fill<<<64, 256, 0, stream>>>(den, 0.0f, (size_t)N_NODES * H);
        k_fill<<<4096, 256, 0, stream>>>(agg, 0.0f, (size_t)N_NODES * HID);
        k_edge_logits<<<4096, 256, 0, stream>>>(qb, kb, eproj, We, src, dst, H, C, exb, mMax);
        k_edge_expsum<<<4096, 256, 0, stream>>>(exb, mMax, dst, den, H);
        k_edge_agg<<<8192, 256, 0, stream>>>(vb, eproj, We, src, dst, exb, den, agg, H, C);

        if (L < 2) k_beta<<<N_NODES, 256, 0, stream>>>(agg, xr, F(bi + 11), t0);
        else       k_add<<<4096, 256, 0, stream>>>(t0, agg, xr, (size_t)N_NODES * HID);

        // LN + GELU (final layer writes straight into hf/hbf)
        k_ln<<<lnB, 256, 0, stream>>>(t0, F(bi + 9), F(bi + 10),
                                      (L == 2) ? hf : t1, hbf, N_NODES, 256, 1);

        if (L < 2) {
            // FFN: GEMM1 (fused GELU, bf16-only out) -> GEMM2 (+bias +residual, f32+bf16 out)
            k_pad_bf16<<<2048, 256, 0, stream>>>(wscr, F(bi + 12), 256, 1024, 256, 1024);
            dim3 gf1((N_NODES + 127) / 128, 1024 / 64);
            k_gemm<<<gf1, 256, 0, stream>>>(hbf, 256, wscr, 1024, N_NODES, F(bi + 13),
                                            nullptr, nullptr, ffbf, 1);
            k_pad_bf16<<<2048, 256, 0, stream>>>(wscr, F(bi + 14), 1024, 256, 1024, 256);
            k_gemm<<<gN, 256, 0, stream>>>(ffbf, 1024, wscr, 256, N_NODES, F(bi + 15),
                                           res, hf, hbf, 0);
        }
    }

    // output head: GEMM(256->128) -> LN+GELU -> dot(128->1)
    k_pad_bf16<<<256, 256, 0, stream>>>(wscr, F(52), 256, 128, 256, 128);
    dim3 go((N_NODES + 127) / 128, 128 / 64);
    k_gemm<<<go, 256, 0, stream>>>(hbf, 256, wscr, 128, N_NODES, F(53), nullptr, t0, nullptr, 0);
    k_ln<<<lnB, 256, 0, stream>>>(t0, F(54), F(55), t1, nullptr, N_NODES, 128, 1);
    k_out<<<(N_NODES * 32 + 255) / 256, 256, 0, stream>>>(t1, F(56), F(57), (float*)d_out);
}
